// LSTMCellFromScratch_47588237639946
// MI455X (gfx1250) — matrix operand order repaired and checked by reading, not compiled, not run on hardware
//
#include <hip/hip_runtime.h>

typedef __attribute__((ext_vector_type(16))) __bf16 v16bf;
typedef __attribute__((ext_vector_type(8)))  float  v8f;
typedef __attribute__((ext_vector_type(4)))  __bf16 bf16x4;

#define BATCH 4096
#define KD    1024   // K per phase (IN == H == 1024)
#define HD    1024

// ---------------- f32 -> bf16 staging ----------------
__global__ __launch_bounds__(256) void cvt_f32_bf16(const float* __restrict__ src,
                                                    __bf16* __restrict__ dst, int n4) {
    int i = blockIdx.x * blockDim.x + threadIdx.x;
    if (i >= n4) return;
    const float4 f = reinterpret_cast<const float4*>(src)[i];
    bf16x4 o;
    o.x = (__bf16)f.x; o.y = (__bf16)f.y; o.z = (__bf16)f.z; o.w = (__bf16)f.w;
    reinterpret_cast<bf16x4*>(dst)[i] = o;
}

// ---------------- fused LSTM GEMM ----------------
union Frag { uint4 u4[2]; v16bf v; };

__device__ __forceinline__ float fast_sigmoid(float x) {
    return 1.0f / (1.0f + __expf(-x));
}
__device__ __forceinline__ float fast_tanh(float x) {
    float e = __expf(2.0f * x);
    return (e - 1.0f) / (e + 1.0f);
}

__global__ __launch_bounds__(256) void lstm_wmma_kernel(
    const __bf16* __restrict__ xb,  const __bf16* __restrict__ hb,
    const __bf16* __restrict__ Wxi, const __bf16* __restrict__ Whi,
    const __bf16* __restrict__ Wxf, const __bf16* __restrict__ Whf,
    const __bf16* __restrict__ Wxg, const __bf16* __restrict__ Whg,
    const __bf16* __restrict__ Wxo, const __bf16* __restrict__ Who,
    const float* __restrict__ b_i, const float* __restrict__ b_f,
    const float* __restrict__ b_g, const float* __restrict__ b_o,
    const float* __restrict__ c_prev, float* __restrict__ out) {

    const int lane = threadIdx.x & 31;
    const int wv   = threadIdx.x >> 5;
    const int half = lane >> 4;    // 0 or 1 (which 16-lane half)
    const int l16  = lane & 15;

    // hidden-unit column this lane contributes to (B rows & epilogue column)
    const int nCol  = blockIdx.x * 128 + wv * 16 + l16;
    const int mBase = blockIdx.y * 64;

    v8f acc[4][4];  // [gate i/f/g/o][M subtile]
    #pragma unroll
    for (int g = 0; g < 4; ++g)
        #pragma unroll
        for (int i = 0; i < 4; ++i)
            acc[g][i] = v8f{};

    // A fragment (16x32 bf16): lanes 0-15 hold K 0-7 & 16-23; lanes 16-31 hold K 8-15 & 24-31
    const int koffA = half * 8;
    // B fragment (32x16 bf16): lanes 0-15 hold K 0-7 & 16-23; lanes 16-31 hold K 8-15 & 24-31
    const int koffB = half * 8;

    #pragma unroll
    for (int ph = 0; ph < 2; ++ph) {
        const __bf16* Ap = ph ? hb : xb;
        const __bf16* Wp[4];
        if (ph == 0) { Wp[0] = Wxi; Wp[1] = Wxf; Wp[2] = Wxg; Wp[3] = Wxo; }
        else         { Wp[0] = Whi; Wp[1] = Whf; Wp[2] = Whg; Wp[3] = Who; }

        const uint4* arow[4];
        #pragma unroll
        for (int i = 0; i < 4; ++i)
            arow[i] = reinterpret_cast<const uint4*>(
                Ap + (size_t)(mBase + i * 16 + l16) * KD + koffA);
        const uint4* brow[4];
        #pragma unroll
        for (int g = 0; g < 4; ++g)
            brow[g] = reinterpret_cast<const uint4*>(
                Wp[g] + (size_t)nCol * KD + koffB);

        for (int kb = 0; kb < KD; kb += 32) {
            const int q = kb >> 3;   // kb in uint4 (8 x bf16) units
            Frag a[4], b[4];
            #pragma unroll
            for (int i = 0; i < 4; ++i) {
                a[i].u4[0] = arow[i][q];       // K = koffA + kb .. +7
                a[i].u4[1] = arow[i][q + 2];   // K = koffA + kb + 16 .. +23
            }
            #pragma unroll
            for (int g = 0; g < 4; ++g) {
                b[g].u4[0] = brow[g][q];       // K = koffB + kb .. +7
                b[g].u4[1] = brow[g][q + 2];   // K = koffB + kb + 16 .. +23
            }
            #pragma unroll
            for (int g = 0; g < 4; ++g)
                #pragma unroll
                for (int i = 0; i < 4; ++i)
                    acc[g][i] = __builtin_amdgcn_wmma_f32_16x16x32_bf16(
                        false, a[i].v, false, b[g].v,
                        (short)0, acc[g][i], false, false);
        }
    }

    // ---------------- fused epilogue ----------------
    const float biv = b_i[nCol];
    const float bfv = b_f[nCol];
    const float bgv = b_g[nCol];
    const float bov = b_o[nCol];
    const int rbase = half * 8;   // C/D layout: lanes 16-31 hold M = 8..15

    #pragma unroll
    for (int i = 0; i < 4; ++i) {
        #pragma unroll
        for (int r = 0; r < 8; ++r) {
            const int row = mBase + i * 16 + rbase + r;
            const size_t idx = (size_t)row * HD + nCol;
            const float iv = fast_sigmoid(acc[0][i][r] + biv);
            const float fv = fast_sigmoid(acc[1][i][r] + bfv);
            const float gv = fast_tanh   (acc[2][i][r] + bgv);
            const float ov = fast_sigmoid(acc[3][i][r] + bov);
            const float c  = fv * c_prev[idx] + iv * gv;
            const float h  = ov * fast_tanh(c);
            out[idx] = h;                              // h_t
            out[(size_t)BATCH * HD + idx] = c;         // c_t
        }
    }
}

extern "C" void kernel_launch(void* const* d_in, const int* in_sizes, int n_in,
                              void* d_out, int out_size, void* d_ws, size_t ws_size,
                              hipStream_t stream) {
    const float* x     = (const float*)d_in[0];
    const float* hprev = (const float*)d_in[1];
    const float* cprev = (const float*)d_in[2];
    const float* W_xi  = (const float*)d_in[3];
    const float* W_hi  = (const float*)d_in[4];
    const float* b_i   = (const float*)d_in[5];
    const float* W_xf  = (const float*)d_in[6];
    const float* W_hf  = (const float*)d_in[7];
    const float* b_f   = (const float*)d_in[8];
    const float* W_xg  = (const float*)d_in[9];
    const float* W_hg  = (const float*)d_in[10];
    const float* b_g   = (const float*)d_in[11];
    const float* W_xo  = (const float*)d_in[12];
    const float* W_ho  = (const float*)d_in[13];
    const float* b_o   = (const float*)d_in[14];

    __bf16* ws = (__bf16*)d_ws;
    const size_t XN = (size_t)BATCH * KD;  // 4M elements
    const size_t WN = (size_t)HD * KD;     // 1M elements

    __bf16* xb = ws;
    __bf16* hb = ws + XN;
    __bf16* wb = ws + 2 * XN;
    __bf16* Wxi_b = wb + 0 * WN;
    __bf16* Whi_b = wb + 1 * WN;
    __bf16* Wxf_b = wb + 2 * WN;
    __bf16* Whf_b = wb + 3 * WN;
    __bf16* Wxg_b = wb + 4 * WN;
    __bf16* Whg_b = wb + 5 * WN;
    __bf16* Wxo_b = wb + 6 * WN;
    __bf16* Who_b = wb + 7 * WN;

    auto cvt = [&](const float* s, __bf16* d, size_t n) {
        int n4 = (int)(n / 4);
        cvt_f32_bf16<<<(n4 + 255) / 256, 256, 0, stream>>>(s, d, n4);
    };
    cvt(x,     xb,    XN);
    cvt(hprev, hb,    XN);
    cvt(W_xi,  Wxi_b, WN);
    cvt(W_hi,  Whi_b, WN);
    cvt(W_xf,  Wxf_b, WN);
    cvt(W_hf,  Whf_b, WN);
    cvt(W_xg,  Wxg_b, WN);
    cvt(W_hg,  Whg_b, WN);
    cvt(W_xo,  Wxo_b, WN);
    cvt(W_ho,  Who_b, WN);

    dim3 grid(HD / 128, BATCH / 64);   // (8, 64)
    lstm_wmma_kernel<<<grid, 256, 0, stream>>>(
        xb, hb,
        Wxi_b, Whi_b, Wxf_b, Whf_b, Wxg_b, Whg_b, Wxo_b, Who_b,
        b_i, b_f, b_g, b_o, cprev, (float*)d_out);
}
